// Attention_4243427688533
// MI455X (gfx1250) — matrix operand order repaired and checked by reading, not compiled, not run on hardware
//
#include <hip/hip_runtime.h>

// ---------------------------------------------------------------------------
// Attention forward for MI455X (gfx1250), wave32 + WMMA f16 (fp32 accum).
// B=4, S=2048, D=1024, H=16, DH=64.
// All GEMM fragments loaded as contiguous 16B runs straight from global
// (weights pre-transposed/converted to f16 once); only the flash-attention
// P-tile relayout uses LDS.
// ---------------------------------------------------------------------------

typedef _Float16 h4  __attribute__((ext_vector_type(4)));
typedef _Float16 h8  __attribute__((ext_vector_type(8)));
typedef _Float16 h16 __attribute__((ext_vector_type(16)));
typedef float    f8  __attribute__((ext_vector_type(8)));

static constexpr int Bc = 4, Sc = 2048, Dc = 1024, Hc = 16, DHc = 64;

// --- helpers ---------------------------------------------------------------

__device__ __forceinline__ h16 pack2(h8 a, h8 b) {
  union { h16 v; h8 h[2]; } u; u.h[0] = a; u.h[1] = b; return u.v;
}
// Two contiguous 16B runs -> one 16-half fragment register set.
__device__ __forceinline__ h16 load2(const _Float16* p0, const _Float16* p1) {
  return pack2(*(const h8*)p0, *(const h8*)p1);
}
__device__ __forceinline__ f8 wmma_f16(h16 a, h16 b, f8 c) {
  // v_wmma_f32_16x16x32_f16  D = A*B + C
  return __builtin_amdgcn_wmma_f32_16x16x32_f16(false, a, false, b,
                                                (short)0, c, false, false);
}
// reductions across the 16-lane half-groups (rows of one C-frag half)
__device__ __forceinline__ float redmax16(float v) {
#pragma unroll
  for (int m = 1; m < 16; m <<= 1) v = fmaxf(v, __shfl_xor(v, m, 32));
  return v;
}
__device__ __forceinline__ float redsum16(float v) {
#pragma unroll
  for (int m = 1; m < 16; m <<= 1) v += __shfl_xor(v, m, 32);
  return v;
}

// --- kernel 0: x fp32 -> f16 ----------------------------------------------
__global__ __launch_bounds__(256) void k_cvt(const float* __restrict__ x,
                                             _Float16* __restrict__ x16, int n4) {
  int i = blockIdx.x * blockDim.x + threadIdx.x;
  if (i < n4) {
    float4 v = ((const float4*)x)[i];
    h4 o; o.x = (_Float16)v.x; o.y = (_Float16)v.y;
    o.z = (_Float16)v.z; o.w = (_Float16)v.w;
    ((h4*)x16)[i] = o;
  }
}

// --- kernel 0b: transpose + convert weights: (R,C) f32 -> (C,R) f16 --------
// grid: (C/32, R/32, nmat)   block: (32, 8)
__global__ __launch_bounds__(256) void k_tr(const float* __restrict__ in,
                                            _Float16* __restrict__ out,
                                            int R, int C) {
  __shared__ float t[32][33];
  const size_t mat = (size_t)blockIdx.z * R * C;
  const float* ip = in + mat;
  _Float16* op = out + mat;
  const int c0 = blockIdx.x * 32, r0 = blockIdx.y * 32;
  const int tx = threadIdx.x, ty = threadIdx.y;
#pragma unroll
  for (int i = 0; i < 32; i += 8)
    t[ty + i][tx] = ip[(size_t)(r0 + ty + i) * C + c0 + tx];
  __syncthreads();
#pragma unroll
  for (int i = 0; i < 32; i += 8)
    op[(size_t)(c0 + ty + i) * R + r0 + tx] = (_Float16)t[tx][ty + i];
}

// --- kernel 1: QKV projections ---------------------------------------------
// grid: (8192/64, H, 3)  block: 256 (8 waves; 4 row-tiles x 2 col-tiles)
// A from x16 (f16), B from pre-transposed WT (H, DH, D) f16 -> all global.
// z==0: Q (prescaled by 1/8)   z==1: K   z==2: V stored transposed (DH x S)
__global__ __launch_bounds__(256) void k_qkv(
    const _Float16* __restrict__ x16,
    const _Float16* __restrict__ WTQ, const _Float16* __restrict__ WTK,
    const _Float16* __restrict__ WTV,
    const float* __restrict__ bQ, const float* __restrict__ bK,
    const float* __restrict__ bV,
    _Float16* __restrict__ Q16, _Float16* __restrict__ K16,
    _Float16* __restrict__ VT16) {
  const int h = blockIdx.y, z = blockIdx.z;
  const _Float16* WT  = (z == 0) ? WTQ : (z == 1) ? WTK : WTV;
  const float*   bias = (z == 0) ? bQ  : (z == 1) ? bK  : bV;
  const _Float16* wt = WT + (size_t)h * DHc * Dc;  // (DH, D) col-of-W layout
  const float* bg = bias + h * DHc;

  const int tid = threadIdx.x;
  const int wv = tid >> 5, lane = tid & 31;
  const int hf = lane >> 4, ln = lane & 15;
  const int m0 = blockIdx.x * 64 + (wv & 3) * 16;
  const int n0 = (wv >> 2) * 32;

  // Per-lane base pointers (k-invariant parts)
  const _Float16* rp0 = x16 + (size_t)(m0 + ln) * Dc + hf * 16;     // A row
  const _Float16* bp0 = wt + (size_t)(n0 + ln) * Dc + hf * 16;      // B col f=0
  const _Float16* bp1 = wt + (size_t)(n0 + 16 + ln) * Dc + hf * 16; // B col f=1

  f8 acc[2] = {};
#pragma unroll 2
  for (int k0 = 0; k0 < Dc; k0 += 32) {
    h16 a  = load2(rp0 + k0, rp0 + k0 + 8);
    h16 b0 = load2(bp0 + k0, bp0 + k0 + 8);
    h16 b1 = load2(bp1 + k0, bp1 + k0 + 8);
    __builtin_prefetch(rp0 + k0 + 64, 0, 1);
    acc[0] = wmma_f16(a, b0, acc[0]);
    acc[1] = wmma_f16(a, b1, acc[1]);
  }

  const float scale = (z == 0) ? 0.125f : 1.0f;  // 1/sqrt(DH) folded into Q
#pragma unroll
  for (int f = 0; f < 2; f++) {
    const int e = n0 + f * 16 + ln;
    const float bv = bg[e];
#pragma unroll
    for (int r = 0; r < 8; r++) {
      const int m = m0 + r + 8 * hf;         // global row in [0, B*S)
      const int b = m >> 11, s = m & (Sc - 1);
      const float v = (acc[f][r] + bv) * scale;
      if (z == 2)
        VT16[((size_t)(b * Hc + h) * DHc + e) * Sc + s] = (_Float16)v;
      else if (z == 0)
        Q16[((size_t)(b * Hc + h) * Sc + s) * DHc + e] = (_Float16)v;
      else
        K16[((size_t)(b * Hc + h) * Sc + s) * DHc + e] = (_Float16)v;
    }
  }
}

// --- kernel 2: causal flash attention --------------------------------------
// grid: (S/64, H, B)  block: 128 (4 waves, 16 queries per wave)
__global__ __launch_bounds__(128) void k_attn(
    const _Float16* __restrict__ Q16, const _Float16* __restrict__ K16,
    const _Float16* __restrict__ VT16, _Float16* __restrict__ Z16) {
  const int h = blockIdx.y, b = blockIdx.z;
  const int tid = threadIdx.x;
  const int wv = tid >> 5, lane = tid & 31;
  const int hf = lane >> 4, ln = lane & 15;
  const int q0 = blockIdx.x * 64 + wv * 16;

  const _Float16* qb = Q16 + (size_t)(b * Hc + h) * Sc * DHc;
  const _Float16* kb = K16 + (size_t)(b * Hc + h) * Sc * DHc;
  const _Float16* vb = VT16 + (size_t)(b * Hc + h) * DHc * Sc;

  __shared__ __align__(16) _Float16 P[4][16][32];  // per-wave prob tile

  // Preload Q A-fragments for the two 32-wide K-dim steps of DH=64.
  const _Float16* qr = qb + (size_t)(q0 + ln) * DHc;
  const h16 aq0 = load2(qr + hf * 16, qr + 8 + hf * 16);
  const h16 aq1 = load2(qr + 32 + hf * 16, qr + 40 + hf * 16);

  f8 acc[4] = {};
  float mrun[8], lrun[8];
#pragma unroll
  for (int r = 0; r < 8; r++) { mrun[r] = -3.0e38f; lrun[r] = 0.0f; }

  const int nch = ((q0 + 15) >> 5) + 1;  // 32-key chunks covering k <= q0+15
  for (int c = 0; c < nch; c++) {
    const int kc = c * 32;
    // ---- S = Q K^T (pre-scaled) : 4 WMMAs ----
    f8 s0 = {}, s1 = {};
    {
      const _Float16* kp0 = kb + (size_t)(kc + ln) * DHc;
      const _Float16* kp1 = kb + (size_t)(kc + 16 + ln) * DHc;
      h16 b00 = load2(kp0 + hf * 16, kp0 + hf * 16 + 8);
      h16 b01 = load2(kp0 + 32 + hf * 16, kp0 + 32 + hf * 16 + 8);
      h16 b10 = load2(kp1 + hf * 16, kp1 + hf * 16 + 8);
      h16 b11 = load2(kp1 + 32 + hf * 16, kp1 + 32 + hf * 16 + 8);
      s0 = wmma_f16(aq0, b00, s0); s0 = wmma_f16(aq1, b01, s0);
      s1 = wmma_f16(aq0, b10, s1); s1 = wmma_f16(aq1, b11, s1);
    }
    // ---- causal mask ----
    const int k0c = kc + ln, k1c = kc + 16 + ln;
#pragma unroll
    for (int r = 0; r < 8; r++) {
      const int q = q0 + r + 8 * hf;
      if (k0c > q) s0[r] = -1.0e30f;
      if (k1c > q) s1[r] = -1.0e30f;
    }
    // ---- online softmax ----
    float mnew[8], corr[8], p0[8], p1[8];
#pragma unroll
    for (int r = 0; r < 8; r++) {
      float cm = redmax16(fmaxf(s0[r], s1[r]));
      mnew[r] = fmaxf(mrun[r], cm);
      corr[r] = __expf(mrun[r] - mnew[r]);
      mrun[r] = mnew[r];
    }
#pragma unroll
    for (int r = 0; r < 8; r++) {
      p0[r] = __expf(s0[r] - mnew[r]);
      p1[r] = __expf(s1[r] - mnew[r]);
      lrun[r] = lrun[r] * corr[r] + redsum16(p0[r] + p1[r]);
    }
#pragma unroll
    for (int j = 0; j < 4; j++)
#pragma unroll
      for (int r = 0; r < 8; r++) acc[j][r] *= corr[r];
    // ---- P: C-frag layout -> LDS -> A-frag layout ----
#pragma unroll
    for (int r = 0; r < 8; r++) {
      P[wv][r + 8 * hf][ln]      = (_Float16)p0[r];
      P[wv][r + 8 * hf][16 + ln] = (_Float16)p1[r];
    }
    asm volatile("s_wait_dscnt 0" ::: "memory");  // intra-wave LDS RAW
    const _Float16* pp = &P[wv][ln][hf * 16];
    h16 ap = load2(pp, pp + 8);
    // ---- acc += P * V : 4 WMMAs (V transposed -> contiguous B-frags) ----
#pragma unroll
    for (int j = 0; j < 4; j++) {
      const _Float16* vp = vb + (size_t)(j * 16 + ln) * Sc + kc + hf * 16;
      h16 bvf = load2(vp, vp + 8);
      acc[j] = wmma_f16(ap, bvf, acc[j]);
    }
  }

  // ---- epilogue: z = acc / l, store f16 (B,S,H*DH) ----
#pragma unroll
  for (int j = 0; j < 4; j++) {
#pragma unroll
    for (int r = 0; r < 8; r++) {
      const int q = q0 + r + 8 * hf;
      const int ec = h * DHc + j * 16 + ln;
      Z16[((size_t)b * Sc + q) * (Hc * DHc) + ec] = (_Float16)(acc[j][r] / lrun[r]);
    }
  }
}

// --- kernel 3: output projection -------------------------------------------
// out(8192x1024) = Z16(8192x1024) * WO(1024x1024) + bO
// B from pre-transposed WOT (d, k) f16.  grid (128,16), 256 thr
__global__ __launch_bounds__(256) void k_oproj(
    const _Float16* __restrict__ Z16, const _Float16* __restrict__ WOT,
    const float* __restrict__ bO, float* __restrict__ out) {
  const int tid = threadIdx.x;
  const int wv = tid >> 5, lane = tid & 31;
  const int hf = lane >> 4, ln = lane & 15;
  const int m0 = blockIdx.x * 64 + (wv & 3) * 16;
  const int d0 = blockIdx.y * 64;
  const int n0 = (wv >> 2) * 32;
  const int HD = Hc * DHc;  // 1024 = K dim

  const _Float16* rp0 = Z16 + (size_t)(m0 + ln) * HD + hf * 16;
  const _Float16* bp0 = WOT + (size_t)(d0 + n0 + ln) * HD + hf * 16;
  const _Float16* bp1 = WOT + (size_t)(d0 + n0 + 16 + ln) * HD + hf * 16;

  f8 acc[2] = {};
#pragma unroll 2
  for (int k0 = 0; k0 < HD; k0 += 32) {
    h16 a  = load2(rp0 + k0, rp0 + k0 + 8);
    h16 b0 = load2(bp0 + k0, bp0 + k0 + 8);
    h16 b1 = load2(bp1 + k0, bp1 + k0 + 8);
    __builtin_prefetch(rp0 + k0 + 64, 0, 1);
    acc[0] = wmma_f16(a, b0, acc[0]);
    acc[1] = wmma_f16(a, b1, acc[1]);
  }
#pragma unroll
  for (int f = 0; f < 2; f++) {
    const int d = d0 + n0 + f * 16 + ln;
    const float bias = bO[d];
#pragma unroll
    for (int r = 0; r < 8; r++) {
      const int m = m0 + r + 8 * hf;
      out[(size_t)m * Dc + d] = acc[f][r] + bias;
    }
  }
}

// ---------------------------------------------------------------------------
extern "C" void kernel_launch(void* const* d_in, const int* in_sizes, int n_in,
                              void* d_out, int out_size, void* d_ws,
                              size_t ws_size, hipStream_t stream) {
  const float* x  = (const float*)d_in[0];
  const float* WQ = (const float*)d_in[1];
  const float* bQ = (const float*)d_in[2];
  const float* WK = (const float*)d_in[3];
  const float* bK = (const float*)d_in[4];
  const float* WV = (const float*)d_in[5];
  const float* bV = (const float*)d_in[6];
  const float* WO = (const float*)d_in[7];
  const float* bO = (const float*)d_in[8];

  char* ws = (char*)d_ws;
  const size_t MB = 1u << 20;
  _Float16* x16  = (_Float16*)(ws + 0 * MB);   // 16 MB (B*S, D) f16
  _Float16* Q16  = (_Float16*)(ws + 16 * MB);  // 16 MB (B,H,S,DH) f16 (prescaled)
  _Float16* K16  = (_Float16*)(ws + 32 * MB);  // 16 MB (B,H,S,DH) f16
  _Float16* VT16 = (_Float16*)(ws + 48 * MB);  // 16 MB (B,H,DH,S) f16
  _Float16* Z16  = (_Float16*)(ws + 64 * MB);  // 16 MB (B,S,H*DH) f16
  _Float16* WTQ  = (_Float16*)(ws + 80 * MB);  // 2 MB (H,DH,D) f16
  _Float16* WTK  = (_Float16*)(ws + 82 * MB);  // 2 MB
  _Float16* WTV  = (_Float16*)(ws + 84 * MB);  // 2 MB
  _Float16* WOT  = (_Float16*)(ws + 86 * MB);  // 2 MB (D, H*DH) f16

  const int n4 = (Bc * Sc * Dc) / 4;  // 2M float4
  k_cvt<<<n4 / 256, 256, 0, stream>>>(x, x16, n4);

  // Transpose+convert weights: per head (D,DH) -> (DH,D); WO (1024,1024) -> T.
  dim3 trb(32, 8);
  k_tr<<<dim3(DHc / 32, Dc / 32, Hc), trb, 0, stream>>>(WQ, WTQ, Dc, DHc);
  k_tr<<<dim3(DHc / 32, Dc / 32, Hc), trb, 0, stream>>>(WK, WTK, Dc, DHc);
  k_tr<<<dim3(DHc / 32, Dc / 32, Hc), trb, 0, stream>>>(WV, WTV, Dc, DHc);
  k_tr<<<dim3(Dc / 32, Dc / 32, 1), trb, 0, stream>>>(WO, WOT, Dc, Dc);

  k_qkv<<<dim3((Bc * Sc) / 64, Hc, 3), 256, 0, stream>>>(
      x16, WTQ, WTK, WTV, bQ, bK, bV, Q16, K16, VT16);

  k_attn<<<dim3(Sc / 64, Hc, Bc), 128, 0, stream>>>(Q16, K16, VT16, Z16);

  k_oproj<<<dim3((Bc * Sc) / 64, Dc / 64), 256, 0, stream>>>(
      Z16, WOT, bO, (float*)d_out);
}
